// GATConv_37452114821313
// MI455X (gfx1250) — matrix operand order repaired and checked by reading, not compiled, not run on hardware
//
#include <hip/hip_runtime.h>
#include <hip/hip_bf16.h>

#define NN   10000
#define EE   160000
#define DD   512
#define HH   8

typedef __attribute__((ext_vector_type(16))) __bf16 v16bf;
typedef __attribute__((ext_vector_type(8)))  float  v8f;

__device__ __forceinline__ __bf16 f2bf(float x) { return (__bf16)x; }

__device__ __forceinline__ void atomicMaxF(float* addr, float val) {
    // monotone int-punning trick (valid with -inf init)
    if (val >= 0.0f) atomicMax((int*)addr, __float_as_int(val));
    else             atomicMin((unsigned int*)addr, __float_as_uint(val));
}

// ---------------------------------------------------------------------------
// WMMA GEMM: C[M x 512] = (A[M x 512] * W[512 x 512] + bias) * scale
// block = 256 threads (8 waves), block tile = 128 (rows) x 128 (cols)
// K-chunks of 32, f32 -> bf16 staged through LDS, v_wmma_f32_16x16x32_bf16
// Each wave: 16 rows x 128 cols = 8 accumulators -> 8 WMMA per K-chunk.
// Tail rows handled by branchless clamp (row r only feeds output row r).
// ---------------------------------------------------------------------------
__global__ void __launch_bounds__(256)
gat_proj_gemm(const float* __restrict__ A, const float* __restrict__ W,
              const float* __restrict__ bias, float* __restrict__ C,
              int M, float scale) {
    __shared__ __bf16 As[128][40];   // [row][k]   stride 40 keeps 16B alignment
    __shared__ __bf16 Bs[128][40];   // [col][k]   (W tile stored transposed)

    const int t       = threadIdx.x;
    const int wave    = t >> 5;
    const int lane    = t & 31;
    const int half    = lane >> 4;   // 0: lanes 0-15, 1: lanes 16-31
    const int mn      = lane & 15;
    const int colBase = blockIdx.x * 128;
    const int rowBase = blockIdx.y * 128;

    v8f acc[8] = {};                 // 8 N-subtiles of 16 per wave

    // staging index precompute
    const int ar = t >> 3;           // 0..31  (A row within 32-row slab)
    const int ak = (t & 7) * 4;      // 0..28  (A k, float4)
    const int wk = t >> 5;           // 0..7   (W k within 8-row slab)
    const int wc = (t & 31) * 4;     // 0..124 (W col, float4)

    for (int kt = 0; kt < 512; kt += 32) {
        // ---- batched global loads (branchless row clamp on A) ----
        float4 av[4], wv[4];
        #pragma unroll
        for (int it = 0; it < 4; ++it) {
            int grow = min(rowBase + it * 32 + ar, M - 1);
            av[it] = *(const float4*)&A[(size_t)grow * 512 + kt + ak];
        }
        #pragma unroll
        for (int it = 0; it < 4; ++it) {
            int k = it * 8 + wk;
            wv[it] = *(const float4*)&W[(size_t)(kt + k) * 512 + colBase + wc];
        }
        if (kt + 32 < 512)
            __builtin_prefetch(&A[(size_t)min(rowBase + ar, M - 1) * 512 + kt + 32 + ak], 0, 1);

        // ---- convert + store to LDS ----
        #pragma unroll
        for (int it = 0; it < 4; ++it) {
            __bf16* d = &As[it * 32 + ar][ak];
            d[0] = f2bf(av[it].x); d[1] = f2bf(av[it].y);
            d[2] = f2bf(av[it].z); d[3] = f2bf(av[it].w);
        }
        #pragma unroll
        for (int it = 0; it < 4; ++it) {
            int k = it * 8 + wk;
            Bs[wc + 0][k] = f2bf(wv[it].x);
            Bs[wc + 1][k] = f2bf(wv[it].y);
            Bs[wc + 2][k] = f2bf(wv[it].z);
            Bs[wc + 3][k] = f2bf(wv[it].w);
        }
        __syncthreads();

        // ---- A fragment: 16x32 bf16, contiguous k order k = half*16 + i (same as B) ----
        v16bf afrag;
        const __bf16* arow = &As[wave * 16 + mn][0];
        #pragma unroll
        for (int i = 0; i < 16; ++i) afrag[i] = arow[half * 16 + i];          // K = half*16 + i

        // ---- 8 B fragments (32x16 each) + WMMA ----
        #pragma unroll
        for (int n = 0; n < 8; ++n) {
            v16bf bfrag;
            const __bf16* bcol = &Bs[n * 16 + mn][0];
            #pragma unroll
            for (int i = 0; i < 16; ++i) bfrag[i] = bcol[half * 16 + i];      // K = half*16 + i
            acc[n] = __builtin_amdgcn_wmma_f32_16x16x32_bf16(
                false, afrag, false, bfrag, (short)0, acc[n], false, false);
        }
        __syncthreads();
    }

    // ---- epilogue: bias + scale, C/D layout: VGPR r -> M = half*8 + r ----
    #pragma unroll
    for (int n = 0; n < 8; ++n) {
        int col = colBase + n * 16 + mn;
        float b = bias[col];
        #pragma unroll
        for (int r = 0; r < 8; ++r) {
            int row = rowBase + wave * 16 + half * 8 + r;
            if (row < M) C[(size_t)row * 512 + col] = (acc[n][r] + b) * scale;
        }
    }
}

// ---------------------------------------------------------------------------
// init: emax = -inf, den = 0, agg = 0
// ---------------------------------------------------------------------------
__global__ void gat_init(float* __restrict__ agg, float* __restrict__ den,
                         float* __restrict__ emax) {
    int i = blockIdx.x * 256 + threadIdx.x;
    if (i < NN * 512) agg[i] = 0.0f;
    if (i < NN * HH) { den[i] = 0.0f; emax[i] = -3.402823466e38f; }
}

// ---------------------------------------------------------------------------
// per-edge QK dots: one wave32 per edge, all 8 heads; atomic-max into emax
// ---------------------------------------------------------------------------
__global__ void __launch_bounds__(256)
gat_escore(const float* __restrict__ fQ, const float* __restrict__ fK,
           const int* __restrict__ src, const int* __restrict__ dst,
           float* __restrict__ ews, float* __restrict__ emax) {
    int e = blockIdx.x * 8 + (threadIdx.x >> 5);
    int lane = threadIdx.x & 31;
    if (e >= EE) return;
    int s = src[e], d = dst[e];
    const float* q = fQ + (size_t)s * 512;
    const float* k = fK + (size_t)d * 512;
    float p[HH];
    #pragma unroll
    for (int h = 0; h < HH; ++h) {
        int b = h * 64 + lane;
        p[h] = q[b] * k[b] + q[b + 32] * k[b + 32];
    }
    #pragma unroll
    for (int off = 16; off > 0; off >>= 1) {
        #pragma unroll
        for (int h = 0; h < HH; ++h) p[h] += __shfl_xor(p[h], off, 32);
    }
    if (lane == 0) {
        #pragma unroll
        for (int h = 0; h < HH; ++h) {
            ews[(size_t)e * HH + h] = p[h];
            atomicMaxF(&emax[(size_t)d * HH + h], p[h]);
        }
    }
}

// ---------------------------------------------------------------------------
// ex = exp(e - emax[dst]); den[dst] += ex
// ---------------------------------------------------------------------------
__global__ void gat_exden(const int* __restrict__ dst, const float* __restrict__ emax,
                          float* __restrict__ ews, float* __restrict__ den) {
    int i = blockIdx.x * 256 + threadIdx.x;
    if (i >= EE * HH) return;
    int e = i >> 3, h = i & 7;
    int d = dst[e];
    float x = __expf(ews[i] - emax[d * HH + h]);
    ews[i] = x;
    atomicAdd(&den[d * HH + h], x);
}

// ---------------------------------------------------------------------------
// agg[dst] += fV[src] * (ex / den[dst] / 1.25): one block per edge
// ---------------------------------------------------------------------------
__global__ void __launch_bounds__(256)
gat_aggregate(const float* __restrict__ fV, const float* __restrict__ ews,
              const float* __restrict__ den, const int* __restrict__ src,
              const int* __restrict__ dst, float* __restrict__ agg) {
    __shared__ float a_sh[HH];
    int e = blockIdx.x;
    int t = threadIdx.x;
    int s = src[e], d = dst[e];
    if (t < HH) a_sh[t] = ews[(size_t)e * HH + t] / (den[(size_t)d * HH + t] * 1.25f);
    __syncthreads();
    const float* v = fV + (size_t)s * 512;
    float*       o = agg + (size_t)d * 512;
    int c0 = t, c1 = t + 256;
    atomicAdd(&o[c0], v[c0] * a_sh[c0 >> 6]);
    atomicAdd(&o[c1], v[c1] * a_sh[c1 >> 6]);
}

// ---------------------------------------------------------------------------
extern "C" void kernel_launch(void* const* d_in, const int* in_sizes, int n_in,
                              void* d_out, int out_size, void* d_ws, size_t ws_size,
                              hipStream_t stream) {
    const float* h_K = (const float*)d_in[0];
    const float* h_Q = (const float*)d_in[1];
    const float* h_V = (const float*)d_in[2];
    const int*   src = (const int*)  d_in[3];
    const int*   dst = (const int*)  d_in[4];
    const float* W_K = (const float*)d_in[5];
    const float* b_K = (const float*)d_in[6];
    const float* W_Q = (const float*)d_in[7];
    const float* b_Q = (const float*)d_in[8];
    const float* W_V = (const float*)d_in[9];
    const float* b_V = (const float*)d_in[10];
    const float* W_O = (const float*)d_in[11];
    const float* b_O = (const float*)d_in[12];
    float* out = (float*)d_out;

    float* ws   = (float*)d_ws;
    float* fK   = ws;                          // N*512
    float* fQ   = ws + (size_t)NN * 512;       // N*512
    float* fV   = ws + (size_t)NN * 512 * 2;   // N*512
    float* agg  = ws + (size_t)NN * 512 * 3;   // N*512
    float* ews  = ws + (size_t)NN * 512 * 4;   // E*8
    float* emax = ews + (size_t)EE * HH;       // N*8
    float* den  = emax + (size_t)NN * HH;      // N*8

    dim3 gGemm(512 / 128, (NN + 127) / 128);   // 4 x 79
    dim3 bGemm(256);

    gat_init<<<(NN * 512 + 255) / 256, 256, 0, stream>>>(agg, den, emax);

    gat_proj_gemm<<<gGemm, bGemm, 0, stream>>>(h_K, W_K, b_K, fK, NN, 1.0f);
    gat_proj_gemm<<<gGemm, bGemm, 0, stream>>>(h_Q, W_Q, b_Q, fQ, NN, 0.125f); // 1/sqrt(DV)
    gat_proj_gemm<<<gGemm, bGemm, 0, stream>>>(h_V, W_V, b_V, fV, NN, 1.0f);

    gat_escore<<<EE / 8, 256, 0, stream>>>(fQ, fK, src, dst, ews, emax);
    gat_exden<<<(EE * HH + 255) / 256, 256, 0, stream>>>(dst, emax, ews, den);
    gat_aggregate<<<EE, 256, 0, stream>>>(fV, ews, den, src, dst, agg);

    gat_proj_gemm<<<gGemm, bGemm, 0, stream>>>(agg, W_O, b_O, out, NN, 1.0f);
}
